// ScaledDotProductAttention_52432960749920
// MI455X (gfx1250) — matrix operand order repaired and checked by reading, not compiled, not run on hardware
//
#include <hip/hip_runtime.h>
#include <math.h>

typedef __attribute__((ext_vector_type(16))) _Float16 v16h;
typedef __attribute__((ext_vector_type(8)))  float    v8f;
typedef __attribute__((ext_vector_type(4)))  float    v4f;

#define SEQ 4096
#define DIM 64

// One workgroup = 4 waves (128 threads) handles a 16-row query tile.
// Dynamic LDS: 16 x 4096 f32 scores/probabilities (256 KB) - fits CDNA5's 320KB WGP LDS.
__global__ __launch_bounds__(128) void attn_wmma_kernel(
    const float* __restrict__ q, const float* __restrict__ k,
    const float* __restrict__ v, float* __restrict__ out,
    float* __restrict__ attn)
{
    extern __shared__ float S[];              // [16][4096] scores -> exp values
    __shared__ float red[128];
    __shared__ float rowmax[16];
    __shared__ float rowinv[16];

    const int wg   = blockIdx.x;              // 0..2047
    const int b    = wg >> 8;                 // batch
    const int q0   = (wg & 255) << 4;         // query tile start row
    const int tid  = threadIdx.x;
    const int lane = tid & 31;
    const int wave = tid >> 5;
    const int mrow = lane & 15;               // A row / B col / C col within tile
    const int hi   = lane >> 4;               // half-wave selector
    const int kb   = hi * 8;                  // unused; A/B fragments use K base hi*16

    const float scale = 0.125f;               // 1/sqrt(64), exact power of two

    // ---- Load Q tile as two f16 A-fragments (d=0..31, d=32..63) ----
    // Fold the softmax scale into Q here (exact: x * 2^-3), so the score
    // WMMA chain needs no per-tile scaling VALU work.
    const float* qbase = q + ((size_t)b * SEQ + q0 + mrow) * DIM;
    v16h a0, a1;
    #pragma unroll
    for (int i = 0; i < 16; ++i) {
        int d = hi * 16 + i;                  // K = hi*16 + 0..15
        a0[i] = (_Float16)(qbase[d] * scale);
        a1[i] = (_Float16)(qbase[d + 32] * scale);
    }

    // ---- Phase 1: scores = (Q*scale) @ K^T, striped over 256 key tiles ----
    const float* kbatch = k + (size_t)b * SEQ * DIM;
    for (int t = wave; t < SEQ / 16; t += 4) {
        int key = t * 16 + mrow;              // B-fragment column n = mrow
        const float* krow = kbatch + (size_t)key * DIM + hi * 16;
        __builtin_prefetch(krow + 4 * 16 * DIM, 0, 1);   // next tile for this wave
        v16h b0, b1;
        #pragma unroll
        for (int i = 0; i < 16; ++i) {        // K = hi*16 + i
            b0[i] = (_Float16)krow[i];        // d = 0..31 slice
            b1[i] = (_Float16)krow[i + 32];   // d = 32..63 slice
        }
        v8f c = {};
        c = __builtin_amdgcn_wmma_f32_16x16x32_f16(false, a0, false, b0,
                                                   (short)0, c, false, false);
        c = __builtin_amdgcn_wmma_f32_16x16x32_f16(false, a1, false, b1,
                                                   (short)0, c, false, false);
        int col = t * 16 + mrow;
        #pragma unroll
        for (int r = 0; r < 8; ++r) {         // C row = r + 8*hi
            S[(r + 8 * hi) * SEQ + col] = c[r];
        }
    }
    __syncthreads();

    // ---- Phase 2: softmax over each 4096-wide row (8 threads per row) ----
    const int row = tid >> 3;                 // 0..15
    const int sub = tid & 7;                  // 0..7
    float* Srow = S + row * SEQ;

    float mx = -INFINITY;
    for (int c = sub; c < SEQ; c += 8) mx = fmaxf(mx, Srow[c]);
    red[tid] = mx;
    __syncthreads();
    if (sub == 0) {
        float m2 = red[tid];
        #pragma unroll
        for (int i = 1; i < 8; ++i) m2 = fmaxf(m2, red[tid + i]);
        rowmax[row] = m2;
    }
    __syncthreads();

    float rmax = rowmax[row];
    float sum = 0.0f;
    for (int c = sub; c < SEQ; c += 8) {
        float e = __expf(Srow[c] - rmax);
        Srow[c] = e;                          // store un-normalized probability
        sum += e;
    }
    red[tid] = sum;
    __syncthreads();
    if (sub == 0) {
        float s2 = 0.0f;
        #pragma unroll
        for (int i = 0; i < 8; ++i) s2 += red[tid + i];
        rowinv[row] = 1.0f / s2;
    }
    __syncthreads();

    // ---- Phase 3: write normalized attention ----
    // Non-temporal 128-bit stores: this 537MB stream must not evict the
    // highly-reused K/V working set (2MB/batch) from the 192MB L2.
    float* attn_base = attn + ((size_t)b * SEQ + q0) * SEQ;
    for (int idx4 = tid; idx4 < 16 * SEQ / 4; idx4 += 128) {
        int idx = idx4 * 4;                   // rows are 4096-aligned, no crossing
        float inv = rowinv[idx >> 12];
        v4f val = *(const v4f*)(&S[idx]);
        val = val * inv;
        __builtin_nontemporal_store(val, (v4f*)(attn_base + idx));
    }

    // ---- Phase 4: out = P @ V ; wave w owns output columns w*16..w*16+15 ----
    // Run the WMMA chain on UN-normalized probabilities (exp values <= 1.0,
    // safe in f16); apply rowinv once per output element at the end. This
    // keeps the inner loop to pure LDS-read + cvt + wmma.
    const float* vbase = v + (size_t)b * SEQ * DIM + wave * 16 + mrow;
    v8f acc = {};
    for (int t = 0; t < SEQ / 32; ++t) {
        // A-fragment: probabilities from LDS (f32 -> f16)
        v16h pa;
        int base = t * 32 + hi * 16;
        #pragma unroll
        for (int i = 0; i < 16; ++i) {
            int kk = base + i;
            pa[i] = (_Float16)S[mrow * SEQ + kk];
        }
        // B-fragment: V[t*32 + hi*16 + i][wave*16 + mrow]
        const float* vp = vbase + (size_t)(t * 32 + hi * 16) * DIM;
        __builtin_prefetch(vp + 32 * DIM, 0, 1);
        v16h vb;
        #pragma unroll
        for (int i = 0; i < 16; ++i) vb[i] = (_Float16)vp[(size_t)i * DIM];
        acc = __builtin_amdgcn_wmma_f32_16x16x32_f16(false, pa, false, vb,
                                                     (short)0, acc, false, false);
    }
    float* obase = out + ((size_t)b * SEQ + q0) * DIM + wave * 16 + mrow;
    #pragma unroll
    for (int r = 0; r < 8; ++r) {
        obase[(size_t)(r + 8 * hi) * DIM] = acc[r] * rowinv[r + 8 * hi];
    }
}

extern "C" void kernel_launch(void* const* d_in, const int* in_sizes, int n_in,
                              void* d_out, int out_size, void* d_ws, size_t ws_size,
                              hipStream_t stream) {
    (void)in_sizes; (void)n_in; (void)d_ws; (void)ws_size; (void)out_size;
    const float* q = (const float*)d_in[0];
    const float* k = (const float*)d_in[1];
    const float* v = (const float*)d_in[2];
    float* out  = (float*)d_out;                          // [8,4096,64] first
    float* attn = out + (size_t)8 * SEQ * DIM;            // then [8,4096,4096]

    dim3 grid(8 * (SEQ / 16));                            // 2048 workgroups
    dim3 block(128);                                      // 4 waves (wave32)
    size_t lds_bytes = (size_t)16 * SEQ * sizeof(float);  // 256 KB dynamic LDS
    hipLaunchKernelGGL(attn_wmma_kernel, grid, block, lds_bytes, stream,
                       q, k, v, out, attn);
}
